// MultiHeadLatentAttention_678604832926
// MI455X (gfx1250) — compile-verified, matrix operand order checked by reading, not run on hardware
//
#include <hip/hip_runtime.h>
#include <hip/hip_bf16.h>
#include <cstdint>

// ---------------------------------------------------------------------------
// CDNA5 (gfx1250) MLA: bf16 WMMA GEMMs (TDM-staged A tiles) + flash attention.
// Wave32; WMMA shape 16x16x32 bf16 -> f32 accumulate.
// ---------------------------------------------------------------------------

typedef __bf16 bf16_t;
typedef bf16_t v16bf __attribute__((ext_vector_type(16)));
typedef bf16_t v8bf  __attribute__((ext_vector_type(8)));
typedef float  v8f   __attribute__((ext_vector_type(8)));
typedef unsigned int u32x4 __attribute__((ext_vector_type(4)));
typedef int i32x8 __attribute__((ext_vector_type(8)));
typedef int i32x4 __attribute__((ext_vector_type(4)));

#if defined(__has_builtin)
#if __has_builtin(__builtin_amdgcn_tensor_load_to_lds) && \
    __has_builtin(__builtin_amdgcn_s_wait_tensorcnt)
#define USE_TDM 1
#endif
#endif
#ifndef USE_TDM
#define USE_TDM 0
#endif

// round-to-nearest-even float -> bf16 (raw u16)
__device__ __forceinline__ unsigned short f2bf(float f) {
  union { float f; unsigned u; } v; v.f = f;
  unsigned u = v.u;
  u += 0x7FFFu + ((u >> 16) & 1u);
  return (unsigned short)(u >> 16);
}

// Build a 16-bit A/B WMMA fragment for one 32-wide K chunk.
// ISA layout (16-bit A-matrix 16x32): lane half hb needs
// K = hb*8 + 0..7 (VGPR0..3) and K = 16 + hb*8 + 0..7 (VGPR4..7).
__device__ __forceinline__ v16bf load_frag(const unsigned short* rowp, int hb) {
  const v8bf lo = *(const v8bf*)(rowp + hb * 8);
  const v8bf hi = *(const v8bf*)(rowp + 16 + hb * 8);
  v16bf r;
#pragma unroll
  for (int j = 0; j < 8; ++j) { r[j] = lo[j]; r[j + 8] = hi[j]; }
  return r;
}

// ---------------------------------------------------------------------------
// GEMM: C = (A[M,K] @ W[K,N] + bias[N]) * alpha
// A fp32 or bf16 (template); W/bias fp32.
// OUT_MODE: 0 = fp32 C[M,N], 1 = bf16 C[M,N], 2 = bf16 transposed Ct[N,M].
// Block tile 128x128, K-step 32, 256 threads = 8 waves (2x4), each wave a
// 64x32 sub-tile = 4x2 WMMA tiles. bf16 A tiles are staged by the TDM.
// ---------------------------------------------------------------------------
template <bool A_BF16, int OUT_MODE>
__global__ __launch_bounds__(256)
void gemm_bias_wmma(const void* __restrict__ Ap, const float* __restrict__ W,
                    const float* __restrict__ bias, void* __restrict__ Cp,
                    int M, int N, int K, float alpha) {
  constexpr int AP = 40;  // LDS row pitch: 32 data + 8 pad bf16 (80B, b128 ok)
  __shared__ __align__(16) unsigned short sA[128 * AP];  // [m][k]
  __shared__ __align__(16) unsigned short sB[128 * AP];  // transposed: [n][k]

  const int tid  = threadIdx.x;
  const int wave = tid >> 5;
  const int lane = tid & 31;
  const int hb   = lane >> 4;
  const int lr   = lane & 15;
  const int wm   = wave >> 2;  // 0..1
  const int wn   = wave & 3;   // 0..3
  const int m0 = blockIdx.y * 128;
  const int n0 = blockIdx.x * 128;

  v8f acc[4][2];
#pragma unroll
  for (int a = 0; a < 4; ++a)
#pragma unroll
    for (int c = 0; c < 2; ++c)
#pragma unroll
      for (int i = 0; i < 8; ++i) acc[a][c][i] = 0.0f;

  for (int k0 = 0; k0 < K; k0 += 32) {
    __syncthreads();
    // ---- stage A tile (128 x 32) ----
    if constexpr (A_BF16) {
#if USE_TDM
      // Tensor Data Mover: 2D tile, 2-byte elements, 128 rows x 32 cols,
      // row stride K, LDS padding 4 dwords after every 16 dwords -> pitch 40.
      if (__builtin_amdgcn_readfirstlane(tid) < 32) {  // wave 0 only (scalar)
        const unsigned short* A16 = (const unsigned short*)Ap;
        const size_t ga = (size_t)(A16 + (size_t)m0 * (size_t)K + (size_t)k0);
        u32x4 g0;
        g0[0] = 1u;                                   // count=1, no gather
        g0[1] = (unsigned)(size_t)(void*)sA;          // lds_addr (bytes)
        g0[2] = (unsigned)(ga & 0xFFFFFFFFu);         // global_addr[31:0]
        g0[3] = (unsigned)((ga >> 32) & 0x01FFFFFFu)  // global_addr[56:32]
                | (2u << 30);                         // type = 2 (image)
        i32x8 g1;
        g1[0] = (int)((1u << 16)      // data_size = 1 -> 2 bytes
                    | (1u << 20)      // pad_enable
                    | (3u << 22)      // pad_interval: 16 dwords
                    | (3u << 25));    // pad_amount: 4 dwords
        g1[1] = (int)(32u << 16);     // tensor_dim0 = 32
        g1[2] = (int)(128u << 16);    // tensor_dim1 = 128
        g1[3] = (int)(32u << 16);     // tile_dim0 = 32
        g1[4] = (int)128u;            // tile_dim1 = 128 (tile_dim2 = 0)
        g1[5] = (int)(unsigned)K;     // tensor_dim0_stride = K
        g1[6] = 0;
        g1[7] = 0;
        const i32x4 gz4 = {0, 0, 0, 0};
        const i32x8 gz8 = {0, 0, 0, 0, 0, 0, 0, 0};
        __builtin_amdgcn_tensor_load_to_lds(g0, g1, gz4, gz4, gz8, 0);
      }
#else
      {
        const int row = tid >> 1;
        const int cb  = (tid & 1) << 4;
        unsigned short* dst = &sA[row * AP + cb];
        const unsigned short* A16 = (const unsigned short*)Ap;
        const size_t gbase = (size_t)(m0 + row) * (size_t)K + (size_t)(k0 + cb);
#pragma unroll
        for (int j = 0; j < 16; ++j) dst[j] = A16[gbase + j];
      }
#endif
    } else {  // fp32 A: convert while staging
      const int row = tid >> 1;
      const int cb  = (tid & 1) << 4;
      unsigned short* dst = &sA[row * AP + cb];
      const float* A32 = (const float*)Ap;
      const size_t gbase = (size_t)(m0 + row) * (size_t)K + (size_t)(k0 + cb);
#pragma unroll
      for (int j = 0; j < 16; ++j) dst[j] = f2bf(A32[gbase + j]);
      if (k0 + 32 < K) __builtin_prefetch(A32 + gbase + 32, 0, 1);
    }
    // ---- stage W tile (32 x 128) transposed into sB[n][k] ----
    {
      const int k  = tid >> 3;        // 0..31
      const int nb = (tid & 7) << 4;  // 0,16,...,112
      const float* Wr = W + (size_t)(k0 + k) * (size_t)N;
#pragma unroll
      for (int j = 0; j < 16; ++j) {
        const int n = n0 + nb + j;
        sB[(nb + j) * AP + k] = (n < N) ? f2bf(Wr[n]) : (unsigned short)0;
      }
      if (k0 + 32 < K) __builtin_prefetch(Wr + (size_t)32 * N + (n0 + nb), 0, 1);
    }
#if USE_TDM
    if constexpr (A_BF16) {
      if (__builtin_amdgcn_readfirstlane(tid) < 32)
        __builtin_amdgcn_s_wait_tensorcnt(0);
    }
#endif
    __syncthreads();

    v16bf afr[4];
#pragma unroll
    for (int mt = 0; mt < 4; ++mt)
      afr[mt] = load_frag(&sA[(wm * 64 + mt * 16 + lr) * AP], hb);
    v16bf bfr[2];
#pragma unroll
    for (int nt = 0; nt < 2; ++nt)
      bfr[nt] = load_frag(&sB[(wn * 32 + nt * 16 + lr) * AP], hb);
#pragma unroll
    for (int mt = 0; mt < 4; ++mt)
#pragma unroll
      for (int nt = 0; nt < 2; ++nt)
        acc[mt][nt] = __builtin_amdgcn_wmma_f32_16x16x32_bf16(
            false, afr[mt], false, bfr[nt], (short)0, acc[mt][nt], false, false);
  }

  // epilogue: D layout -> row = i + 8*hb, col = lr within each 16x16 tile
#pragma unroll
  for (int mt = 0; mt < 4; ++mt) {
#pragma unroll
    for (int nt = 0; nt < 2; ++nt) {
      const int col = n0 + wn * 32 + nt * 16 + lr;
      if (col < N) {
        const float bv = bias[col];
#pragma unroll
        for (int i = 0; i < 8; ++i) {
          const int row = m0 + wm * 64 + mt * 16 + i + 8 * hb;
          const float vv = (acc[mt][nt][i] + bv) * alpha;
          if constexpr (OUT_MODE == 0)
            ((float*)Cp)[(size_t)row * N + col] = vv;
          else if constexpr (OUT_MODE == 1)
            ((unsigned short*)Cp)[(size_t)row * N + col] = f2bf(vv);
          else  // transposed bf16: Ct[col][row]
            ((unsigned short*)Cp)[(size_t)col * (size_t)M + row] = f2bf(vv);
        }
      }
    }
  }
}

// ---------------------------------------------------------------------------
// Flash attention. One block = (batch b, head h, 64-row q tile); 4 waves,
// each wave owns 16 q rows, fully independent (no block barriers).
// Q/K dims: 128 content + 64 rope (rope K shared across heads). V dim 128,
// consumed from the transposed layout Vt[h*128+d][b*1024+l]. Q pre-scaled.
// ---------------------------------------------------------------------------
__global__ __launch_bounds__(128)
void mla_flash_attn(const unsigned short* __restrict__ Qc,
                    const unsigned short* __restrict__ Qr,
                    const unsigned short* __restrict__ Kc,
                    const unsigned short* __restrict__ Kr,
                    const unsigned short* __restrict__ Vt,
                    unsigned short* __restrict__ Ob) {
  constexpr int L = 1024, CD = 8192, RD = 4096, KRD = 64, ROWS = 4096;
  constexpr int VP = 72;  // LDS pitch for P tiles
  __shared__ __align__(16) unsigned short sP[4][16 * VP];  // per-wave P tile

  const int tid  = threadIdx.x;
  const int wave = tid >> 5;
  const int lane = tid & 31;
  const int hb   = lane >> 4;
  const int lr   = lane & 15;

  const int blk = blockIdx.x;
  const int qt = blk & 15;         // q tile (64 rows)
  const int h  = (blk >> 4) & 63;  // head
  const int b  = blk >> 10;        // batch
  const int q0 = qt * 64;

  // Q fragments: 6 K-chunks of 32 (4 content + 2 rope), kept in registers.
  v16bf qf[6];
  {
    const size_t qrow = (size_t)(b * L + q0 + wave * 16 + lr);
    const unsigned short* qcp = Qc + qrow * CD + h * 128;
#pragma unroll
    for (int c = 0; c < 4; ++c) qf[c] = load_frag(qcp + c * 32, hb);
    const unsigned short* qrp = Qr + qrow * RD + h * 64;
#pragma unroll
    for (int c = 0; c < 2; ++c) qf[4 + c] = load_frag(qrp + c * 32, hb);
  }

  v8f o[8];
#pragma unroll
  for (int v = 0; v < 8; ++v)
#pragma unroll
    for (int i = 0; i < 8; ++i) o[v][i] = 0.0f;
  float mrow[8], lrow[8];
#pragma unroll
  for (int i = 0; i < 8; ++i) { mrow[i] = -3.0e38f; lrow[i] = 0.0f; }

  const unsigned short* vbase =
      Vt + (size_t)(h * 128) * ROWS + (size_t)(b * L);  // + d*ROWS + key

  for (int kt = 0; kt <= qt; ++kt) {
    const int k0 = kt * 64;

    // S = Q . K^T  (16 q-rows x 64 keys per wave)
    v8f s[4];
#pragma unroll
    for (int nt = 0; nt < 4; ++nt) {
#pragma unroll
      for (int i = 0; i < 8; ++i) s[nt][i] = 0.0f;
      const int key = k0 + nt * 16 + lr;
      const unsigned short* kcp = Kc + (size_t)(b * L + key) * CD + h * 128;
#pragma unroll
      for (int c = 0; c < 4; ++c) {
        const v16bf kf = load_frag(kcp + c * 32, hb);
        s[nt] = __builtin_amdgcn_wmma_f32_16x16x32_bf16(
            false, qf[c], false, kf, (short)0, s[nt], false, false);
      }
      const unsigned short* krp = Kr + (size_t)(b * L + key) * KRD;
#pragma unroll
      for (int c = 0; c < 2; ++c) {
        const v16bf kf = load_frag(krp + c * 32, hb);
        s[nt] = __builtin_amdgcn_wmma_f32_16x16x32_bf16(
            false, qf[4 + c], false, kf, (short)0, s[nt], false, false);
      }
    }

    if (kt == qt) {  // causal mask on the diagonal tile
#pragma unroll
      for (int nt = 0; nt < 4; ++nt)
#pragma unroll
        for (int i = 0; i < 8; ++i) {
          const int key = nt * 16 + lr;
          const int qq  = wave * 16 + i + 8 * hb;
          if (key > qq) s[nt][i] = -3.0e38f;
        }
    }

    // online softmax; row r = i + 8*hb lives in vgpr i across a 16-lane half
#pragma unroll
    for (int i = 0; i < 8; ++i) {
      float mx = s[0][i];
#pragma unroll
      for (int nt = 1; nt < 4; ++nt) mx = fmaxf(mx, s[nt][i]);
#pragma unroll
      for (int d = 1; d < 16; d <<= 1) mx = fmaxf(mx, __shfl_xor(mx, d, 32));
      const float mnew = fmaxf(mrow[i], mx);
      const float corr = __expf(mrow[i] - mnew);
      mrow[i] = mnew;
      float sum = 0.0f;
#pragma unroll
      for (int nt = 0; nt < 4; ++nt) {
        const float p = __expf(s[nt][i] - mnew);
        s[nt][i] = p;
        sum += p;
      }
#pragma unroll
      for (int d = 1; d < 16; d <<= 1) sum += __shfl_xor(sum, d, 32);
      lrow[i] = lrow[i] * corr + sum;
#pragma unroll
      for (int v = 0; v < 8; ++v) o[v][i] *= corr;
    }

    // P (D layout) -> per-wave LDS tile (row-contiguous), then A fragments
#pragma unroll
    for (int nt = 0; nt < 4; ++nt)
#pragma unroll
      for (int i = 0; i < 8; ++i)
        sP[wave][(i + 8 * hb) * VP + nt * 16 + lr] = f2bf(s[nt][i]);

    v16bf pf[2];
#pragma unroll
    for (int c = 0; c < 2; ++c)
      pf[c] = load_frag(&sP[wave][lr * VP + c * 32], hb);

    // O += P . V   (B fragments straight from transposed V in global)
#pragma unroll
    for (int nt = 0; nt < 8; ++nt) {
      const unsigned short* vp = vbase + (size_t)(nt * 16 + lr) * ROWS + k0;
#pragma unroll
      for (int c = 0; c < 2; ++c) {
        const v16bf vf = load_frag(vp + c * 32, hb);
        o[nt] = __builtin_amdgcn_wmma_f32_16x16x32_bf16(
            false, pf[c], false, vf, (short)0, o[nt], false, false);
      }
    }
  }

  // normalize and store (bf16)
  {
    const size_t rb = (size_t)(b * L + q0 + wave * 16);
#pragma unroll
    for (int nt = 0; nt < 8; ++nt)
#pragma unroll
      for (int i = 0; i < 8; ++i) {
        const float val = o[nt][i] / lrow[i];
        Ob[(rb + i + 8 * hb) * CD + h * 128 + nt * 16 + lr] = f2bf(val);
      }
  }
}

// ---------------------------------------------------------------------------
// Host-side launch.
// ---------------------------------------------------------------------------
extern "C" void kernel_launch(void* const* d_in, const int* in_sizes, int n_in,
                              void* d_out, int out_size, void* d_ws,
                              size_t ws_size, hipStream_t stream) {
  (void)in_sizes; (void)n_in; (void)out_size; (void)ws_size;

  const float* x     = (const float*)d_in[0];
  const float* W_dq  = (const float*)d_in[1];
  const float* b_dq  = (const float*)d_in[2];
  const float* W_dkv = (const float*)d_in[3];
  const float* b_dkv = (const float*)d_in[4];
  const float* W_uq  = (const float*)d_in[5];
  const float* b_uq  = (const float*)d_in[6];
  const float* W_uk  = (const float*)d_in[7];
  const float* b_uk  = (const float*)d_in[8];
  const float* W_uv  = (const float*)d_in[9];
  const float* b_uv  = (const float*)d_in[10];
  const float* W_pq  = (const float*)d_in[11];
  const float* b_pq  = (const float*)d_in[12];
  const float* W_pk  = (const float*)d_in[13];
  const float* b_pk  = (const float*)d_in[14];
  const float* W_o   = (const float*)d_in[15];
  const float* b_o   = (const float*)d_in[16];

  constexpr int ROWS = 4096;  // B * L
  unsigned short* ws = (unsigned short*)d_ws;
  size_t off = 0;
  unsigned short* Qlat  = ws + off; off += (size_t)ROWS * 1536;
  unsigned short* KVlat = ws + off; off += (size_t)ROWS * 512;
  unsigned short* Qc    = ws + off; off += (size_t)ROWS * 8192;
  unsigned short* Qr    = ws + off; off += (size_t)ROWS * 4096;
  unsigned short* Kc    = ws + off; off += (size_t)ROWS * 8192;
  unsigned short* Kr    = ws + off; off += (size_t)ROWS * 64;
  unsigned short* Vt    = ws + off; off += (size_t)ROWS * 8192;  // [N=8192][M=4096]
  unsigned short* Ao    = ws + off; off += (size_t)ROWS * 8192;

  const float alpha_q = 0.08838834764831845f;  // 1/sqrt(128), fused into Q
  const dim3 thr(256);
  auto grid = [](int N) { return dim3((unsigned)((N + 127) / 128), ROWS / 128); };

  // latents (fp32 A)
  gemm_bias_wmma<false, 1><<<grid(1536), thr, 0, stream>>>(
      x, W_dq, b_dq, Qlat, ROWS, 1536, 8192, 1.0f);
  gemm_bias_wmma<false, 1><<<grid(512), thr, 0, stream>>>(
      x, W_dkv, b_dkv, KVlat, ROWS, 512, 8192, 1.0f);
  // up-projections (bf16 A via TDM; Q pre-scaled by 1/sqrt(128))
  gemm_bias_wmma<true, 1><<<grid(8192), thr, 0, stream>>>(
      Qlat, W_uq, b_uq, Qc, ROWS, 8192, 1536, alpha_q);
  gemm_bias_wmma<true, 1><<<grid(4096), thr, 0, stream>>>(
      Qlat, W_pq, b_pq, Qr, ROWS, 4096, 1536, alpha_q);
  gemm_bias_wmma<true, 1><<<grid(8192), thr, 0, stream>>>(
      KVlat, W_uk, b_uk, Kc, ROWS, 8192, 512, 1.0f);
  gemm_bias_wmma<true, 2><<<grid(8192), thr, 0, stream>>>(
      KVlat, W_uv, b_uv, Vt, ROWS, 8192, 512, 1.0f);  // transposed V
  gemm_bias_wmma<false, 1><<<grid(64), thr, 0, stream>>>(
      x, W_pk, b_pk, Kr, ROWS, 64, 8192, 1.0f);

  // attention: B * NH * (L/64) = 4 * 64 * 16 = 4096 blocks
  mla_flash_attn<<<dim3(4096), dim3(128), 0, stream>>>(Qc, Qr, Kc, Kr, Vt, Ao);

  // output projection -> fp32 d_out
  gemm_bias_wmma<true, 0><<<grid(8192), thr, 0, stream>>>(
      Ao, W_o, b_o, d_out, ROWS, 8192, 8192, 1.0f);
}
